// InfoNCELossTriton_27762668601806
// MI455X (gfx1250) — matrix operand order repaired and checked by reading, not compiled, not run on hardware
//
#include <hip/hip_runtime.h>
#include <hip/hip_bf16.h>

typedef __attribute__((ext_vector_type(16))) __bf16 v16bf;
typedef __attribute__((ext_vector_type(8)))  float  v8f;

#define D_DIM 128
#define B_ROWS 512
#define K_ROWS 65536
#define INV_T 5.0f
#define EPS_N 1e-12f

// xor-lane reduction step via ds_swizzle (group-of-32 mode: imm = xor<<10 | and_mask 0x1f)
#define SWZ_XOR_ADD(x, m) \
    ((x) + __int_as_float(__builtin_amdgcn_ds_swizzle(__float_as_int(x), (((m) << 10) | 0x1f))))

// ---------------------------------------------------------------------------
// Kernel A: normalize q and k rows, emit bf16 hi/lo of qn, l_pos, zero sumexp
// One block (128 threads) per row b.
// ---------------------------------------------------------------------------
__global__ __launch_bounds__(128) void prep_qk(
    const float* __restrict__ q, const float* __restrict__ k,
    __bf16* __restrict__ qnhi, __bf16* __restrict__ qnlo,
    float* __restrict__ l_pos, float* __restrict__ row_sumexp)
{
    const int b = blockIdx.x;
    const int t = threadIdx.x;
    const float qv = q[b * D_DIM + t];
    const float kv = k[b * D_DIM + t];

    __shared__ float red[128];

    // sum(q^2)
    red[t] = qv * qv; __syncthreads();
    for (int s = 64; s > 0; s >>= 1) { if (t < s) red[t] += red[t + s]; __syncthreads(); }
    const float qq = red[0]; __syncthreads();

    // sum(k^2)
    red[t] = kv * kv; __syncthreads();
    for (int s = 64; s > 0; s >>= 1) { if (t < s) red[t] += red[t + s]; __syncthreads(); }
    const float kk = red[0]; __syncthreads();

    const float rq = 1.0f / fmaxf(sqrtf(qq), EPS_N);
    const float rk = 1.0f / fmaxf(sqrtf(kk), EPS_N);
    const float qn = qv * rq;
    const float kn = kv * rk;

    // bf16 hi/lo split of qn (compensated precision for the WMMA GEMM)
    const __bf16 h = (__bf16)qn;
    qnhi[b * D_DIM + t] = h;
    qnlo[b * D_DIM + t] = (__bf16)(qn - (float)h);

    // l_pos = dot(qn, kn) * (1/T)
    red[t] = qn * kn; __syncthreads();
    for (int s = 64; s > 0; s >>= 1) { if (t < s) red[t] += red[t + s]; __syncthreads(); }
    if (t == 0) {
        l_pos[b] = red[0] * INV_T;
        row_sumexp[b] = 0.0f;   // re-init every call (graph replay, poisoned ws)
    }
}

// ---------------------------------------------------------------------------
// Kernel B: 1/||queue_j|| for all 65536 rows. One wave per row.
// ---------------------------------------------------------------------------
__global__ __launch_bounds__(256) void prep_queue_norm(
    const float* __restrict__ queue, float* __restrict__ invn)
{
    const int wave = threadIdx.x >> 5;
    const int lane = threadIdx.x & 31;
    const int j = blockIdx.x * 8 + wave;

    const float4* row = reinterpret_cast<const float4*>(queue + (size_t)j * D_DIM);
    const float4 f = row[lane];                 // 32 lanes * 4 = 128 floats
    float ss = f.x * f.x + f.y * f.y + f.z * f.z + f.w * f.w;
    ss = SWZ_XOR_ADD(ss, 1);
    ss = SWZ_XOR_ADD(ss, 2);
    ss = SWZ_XOR_ADD(ss, 4);
    ss = SWZ_XOR_ADD(ss, 8);
    ss = SWZ_XOR_ADD(ss, 16);
    if (lane == 0) invn[j] = 1.0f / fmaxf(sqrtf(ss), EPS_N);
}

// ---------------------------------------------------------------------------
// Kernel C: fused GEMM (bf16x3 compensated WMMA) + exp + per-row sum.
// Grid: (N/128, M/128). Block: 256 threads = 8 waves; wave w owns columns
// n0 + 16*w and ALL 8 M-tiles of the block (8x v8f accumulators), so the
// B-side normalize+bf16-split is amortized over 24 WMMAs per K-chunk and
// the independent C tiles hide WMMA hazard latency.
// K = 128 -> 4 chunks of 32; 3 WMMAs (hi*hi + hi*lo + lo*hi) per tile/chunk.
// ---------------------------------------------------------------------------
__global__ __launch_bounds__(256) void gemm_lse(
    const float* __restrict__ queue,
    const unsigned int* __restrict__ qnhi_u32,   // bf16 pairs
    const unsigned int* __restrict__ qnlo_u32,
    const float* __restrict__ invn,
    float* __restrict__ row_sumexp)
{
    const int wave = threadIdx.x >> 5;
    const int lane = threadIdx.x & 31;
    const int hf   = lane >> 4;        // lane half: selects K sub-range
    const int mr   = lane & 15;        // row (A) / column (B) within tile
    const int m0   = blockIdx.y * 128;
    const int n0   = blockIdx.x * 128;
    const int j    = n0 + wave * 16 + mr;        // queue row this lane feeds

    const float scl = invn[j];

    union V16 { v16bf v; unsigned int u[8]; __bf16 e[16]; };

    v8f c[8] = {};

    // per-lane A K-pair offsets (lane half -> contiguous 16 K values, matches B)
    int kpair[8];
#pragma unroll
    for (int v = 0; v < 8; ++v)
        kpair[v] = 2 * v + hf * 16;

#pragma unroll
    for (int kc = 0; kc < 4; ++kc) {
        const int K0 = kc * 32;

        // ---- B fragment (32x16 bf16): lane half -> contiguous 16 K values ----
        const float* qbase = queue + (size_t)j * D_DIM + K0 + hf * 16;
        if (kc < 3)
            __builtin_prefetch(qbase + 32, 0, 3);   // next K-chunk of this row
        const float4* qrow = reinterpret_cast<const float4*>(qbase);
        V16 bhi, blo;
#pragma unroll
        for (int p = 0; p < 4; ++p) {
            const float4 f = qrow[p];
            const float x0 = f.x * scl, x1 = f.y * scl, x2 = f.z * scl, x3 = f.w * scl;
            const __bf16 h0 = (__bf16)x0, h1 = (__bf16)x1, h2 = (__bf16)x2, h3 = (__bf16)x3;
            bhi.e[4 * p + 0] = h0; bhi.e[4 * p + 1] = h1;
            bhi.e[4 * p + 2] = h2; bhi.e[4 * p + 3] = h3;
            blo.e[4 * p + 0] = (__bf16)(x0 - (float)h0);
            blo.e[4 * p + 1] = (__bf16)(x1 - (float)h1);
            blo.e[4 * p + 2] = (__bf16)(x2 - (float)h2);
            blo.e[4 * p + 3] = (__bf16)(x3 - (float)h3);
        }

        // ---- 8 M-tiles reuse this B fragment ----
#pragma unroll
        for (int mt = 0; mt < 8; ++mt) {
            V16 ahi, alo;
            const int rowbase = (m0 + mt * 16 + mr) * (D_DIM / 2);
#pragma unroll
            for (int v = 0; v < 8; ++v) {
                const int idx = rowbase + ((K0 + kpair[v]) >> 1);
                ahi.u[v] = qnhi_u32[idx];
                alo.u[v] = qnlo_u32[idx];
            }
            // bf16x3 compensated product: hi*hi + hi*lo + lo*hi (f32 accumulate)
            c[mt] = __builtin_amdgcn_wmma_f32_16x16x32_bf16(false, ahi.v, false, bhi.v,
                                                            (short)0, c[mt], false, false);
            c[mt] = __builtin_amdgcn_wmma_f32_16x16x32_bf16(false, ahi.v, false, blo.v,
                                                            (short)0, c[mt], false, false);
            c[mt] = __builtin_amdgcn_wmma_f32_16x16x32_bf16(false, alo.v, false, bhi.v,
                                                            (short)0, c[mt], false, false);
        }
    }

    // ---- epilogue: logits bounded by 1/T=5 (unit vectors) -> max-free exp-sum
    __shared__ float red[8][128];
#pragma unroll
    for (int mt = 0; mt < 8; ++mt) {
#pragma unroll
        for (int r = 0; r < 8; ++r) {
            // C layout: VGPR r -> row M = mt*16 + r + 8*hf, column N = mr
            float e = __expf(c[mt][r] * INV_T);
            e = SWZ_XOR_ADD(e, 1);
            e = SWZ_XOR_ADD(e, 2);
            e = SWZ_XOR_ADD(e, 4);
            e = SWZ_XOR_ADD(e, 8);        // sum over 16 columns (stays in half)
            if (mr == 0) red[wave][mt * 16 + hf * 8 + r] = e;
        }
    }
    __syncthreads();

    if (threadIdx.x < 128) {              // one atomic per row per block
        float t = 0.0f;
#pragma unroll
        for (int w = 0; w < 8; ++w) t += red[w][threadIdx.x];
        atomicAdd(&row_sumexp[m0 + threadIdx.x], t);
    }
}

// ---------------------------------------------------------------------------
// Kernel D: loss = mean_b( log(sumexp_neg + exp(l_pos)) - l_pos )
// ---------------------------------------------------------------------------
__global__ __launch_bounds__(512) void finalize(
    const float* __restrict__ l_pos, const float* __restrict__ row_sumexp,
    float* __restrict__ out)
{
    const int t = threadIdx.x;
    const float lp = l_pos[t];
    const float s  = row_sumexp[t] + __expf(lp);
    float v = __logf(s) - lp;

    __shared__ float red[512];
    red[t] = v; __syncthreads();
    for (int s2 = 256; s2 > 0; s2 >>= 1) {
        if (t < s2) red[t] += red[t + s2];
        __syncthreads();
    }
    if (t == 0) out[0] = red[0] * (1.0f / (float)B_ROWS);
}

// ---------------------------------------------------------------------------
extern "C" void kernel_launch(void* const* d_in, const int* in_sizes, int n_in,
                              void* d_out, int out_size, void* d_ws, size_t ws_size,
                              hipStream_t stream)
{
    (void)in_sizes; (void)n_in; (void)out_size; (void)ws_size;

    const float* q     = (const float*)d_in[0];
    const float* k     = (const float*)d_in[1];
    const float* queue = (const float*)d_in[2];

    char* ws = (char*)d_ws;
    __bf16* qnhi  = (__bf16*)(ws);                 // 512*128*2 = 128 KB
    __bf16* qnlo  = (__bf16*)(ws + (131072));      // 128 KB
    float*  invn  = (float*)(ws + (262144));       // 65536*4 = 256 KB
    float*  lpos  = (float*)(ws + (524288));       // 2 KB
    float*  rowse = (float*)(ws + (526336));       // 2 KB
    float*  out   = (float*)d_out;

    prep_qk<<<B_ROWS, 128, 0, stream>>>(q, k, qnhi, qnlo, lpos, rowse);
    prep_queue_norm<<<K_ROWS / 8, 256, 0, stream>>>(queue, invn);

    dim3 grid(K_ROWS / 128, B_ROWS / 128);
    gemm_lse<<<grid, 256, 0, stream>>>(queue,
                                       (const unsigned int*)qnhi,
                                       (const unsigned int*)qnlo,
                                       invn, rowse);

    finalize<<<1, 512, 0, stream>>>(lpos, rowse, out);
}
